// GCN_51943334477917
// MI455X (gfx1250) — matrix operand order repaired and checked by reading, not compiled, not run on hardware
//
#include <hip/hip_runtime.h>

typedef __attribute__((ext_vector_type(16))) _Float16 v16h;
typedef __attribute__((ext_vector_type(8)))  _Float16 h8;
typedef __attribute__((ext_vector_type(8)))  float    v8f;
typedef __attribute__((ext_vector_type(4)))  float    f4;
typedef __attribute__((ext_vector_type(2)))  float    f2;

// ---------------------------------------------------------------------------
// Pre-pass: split fp32 -> (hi, lo) f16 pair, packed row-major. 8 elems/thread.
// n must be a multiple of 8 (N*128, 128*Fout all are).
// ---------------------------------------------------------------------------
__global__ void split_f16x8(const float* __restrict__ src,
                            _Float16* __restrict__ hi,
                            _Float16* __restrict__ lo, long long n) {
    long long base = ((long long)blockIdx.x * blockDim.x + threadIdx.x) * 8;
    if (base >= n) return;
    h8 h, l;
#pragma unroll
    for (int j = 0; j < 8; ++j) {
        float v = src[base + j];
        _Float16 hh = (_Float16)v;
        h[j] = hh;
        l[j] = (_Float16)(v - (float)hh);
    }
    *(h8*)(hi + base) = h;
    *(h8*)(lo + base) = l;
}

// ---------------------------------------------------------------------------
// Pre-pass: W[K x Fout] -> transposed split-f16 Wt[Fout x K] (hi, lo).
// Tiny matrix (<=16K elems); scalar gather is fine.
// ---------------------------------------------------------------------------
__global__ void wt_split_f16(const float* __restrict__ W,
                             _Float16* __restrict__ wh,
                             _Float16* __restrict__ wl, int K, int Fout) {
    int i = blockIdx.x * blockDim.x + threadIdx.x;
    if (i >= K * Fout) return;
    int f = i / K;
    int k = i - f * K;
    float v = W[(size_t)k * Fout + f];
    _Float16 hh = (_Float16)v;
    wh[i] = hh;
    wl[i] = (_Float16)(v - (float)hh);
}

// ---------------------------------------------------------------------------
// GEMM: H[N x Fout] = X[N x 128] * W[128 x Fout] via split-f16 WMMA
// (hi*hi + hi*lo + lo*hi, fp32 accumulate ~ fp32 accuracy).
// Inputs pre-converted: Xh/Xl row-major [N x 128] f16, Wh/Wl transposed
// [Fout x 128] f16. One wave per 16x16 tile; K=128 fully unrolled -> 12 WMMA.
// ---------------------------------------------------------------------------
__global__ __launch_bounds__(32)
void gcn_gemm_wmma(const _Float16* __restrict__ Xh, const _Float16* __restrict__ Xl,
                   const _Float16* __restrict__ Wh, const _Float16* __restrict__ Wl,
                   float* __restrict__ H, int Fout) {
    const int  lane   = threadIdx.x;          // wave32
    const int  mTile  = blockIdx.x;
    const int  nTile  = blockIdx.y;
    const bool hiHalf = lane >= 16;
    const int  l15    = lane & 15;
    const int  K      = 128;

    const int row = mTile * 16 + l15;
    const int col = nTile * 16 + l15;

    const _Float16* __restrict__ xh = Xh + (size_t)row * K;
    const _Float16* __restrict__ xl = Xl + (size_t)row * K;
    const _Float16* __restrict__ wh = Wh + (size_t)col * K;
    const _Float16* __restrict__ wl = Wl + (size_t)col * K;

    __builtin_prefetch(xh, 0, 1);             // global_prefetch_b8
    __builtin_prefetch(wh, 0, 1);

    const int aoff = hiHalf ? 16 : 0;         // A: lanes 16-31 hold K+16..
    const int boff = hiHalf ? 16 : 0;         // B: lanes 16-31 hold K+16..

    v8f c = {};
#pragma unroll
    for (int k0 = 0; k0 < K; k0 += 32) {
        // A fragment: 8 halves at k0+aoff, 8 at k0+8+aoff (16B loads)
        h8 a0 = *(const h8*)(xh + k0 + aoff);
        h8 a1 = *(const h8*)(xh + k0 + 8 + aoff);
        v16h ah = __builtin_shufflevector(a0, a1, 0,1,2,3,4,5,6,7,
                                                  8,9,10,11,12,13,14,15);
        h8 a2 = *(const h8*)(xl + k0 + aoff);
        h8 a3 = *(const h8*)(xl + k0 + 8 + aoff);
        v16h al = __builtin_shufflevector(a2, a3, 0,1,2,3,4,5,6,7,
                                                  8,9,10,11,12,13,14,15);
        // B fragment: 16 contiguous halves (32B)
        v16h bh = *(const v16h*)(wh + k0 + boff);
        v16h bl = *(const v16h*)(wl + k0 + boff);

        c = __builtin_amdgcn_wmma_f32_16x16x32_f16(false, ah, false, bh,
                                                   (short)0, c, false, false);
        c = __builtin_amdgcn_wmma_f32_16x16x32_f16(false, ah, false, bl,
                                                   (short)0, c, false, false);
        c = __builtin_amdgcn_wmma_f32_16x16x32_f16(false, al, false, bh,
                                                   (short)0, c, false, false);
    }

    // C layout: VGPR v -> row mTile*16 + v (+8 for lanes 16-31), col = l15
    const int mBase = mTile * 16 + (hiHalf ? 8 : 0);
#pragma unroll
    for (int v = 0; v < 8; ++v) {
        H[(size_t)(mBase + v) * Fout + col] = c[v];
    }
}

// ---------------------------------------------------------------------------
// Utility: zero a float buffer
// ---------------------------------------------------------------------------
__global__ void zero_f32(float* __restrict__ p, long long n) {
    long long i = (long long)blockIdx.x * blockDim.x + threadIdx.x;
    if (i < n) p[i] = 0.0f;
}

// ---------------------------------------------------------------------------
// Degree count over target nodes (col); dis[i] = rsqrt(deg[i] + 1) (self-loop)
// ---------------------------------------------------------------------------
__global__ void deg_count(const long long* __restrict__ cols,
                          float* __restrict__ deg, long long E) {
    long long e = (long long)blockIdx.x * blockDim.x + threadIdx.x;
    if (e < E) atomicAdd(&deg[cols[e]], 1.0f);
}

__global__ void deg_finalize(float* __restrict__ deg, int N) {
    int i = blockIdx.x * blockDim.x + threadIdx.x;
    if (i < N) deg[i] = rsqrtf(deg[i] + 1.0f);
}

// ---------------------------------------------------------------------------
// Edge scatter: Acc[col] += dis[row]*dis[col] * Hsrc[row]. One wave32 / edge,
// vectorized gather; fp32 atomics resolve in L2 (dest fits in 192MB L2).
// ---------------------------------------------------------------------------
__global__ void scatter_edges(const long long* __restrict__ rows,
                              const long long* __restrict__ cols,
                              const float* __restrict__ dis,
                              const float* __restrict__ Hsrc,
                              float* __restrict__ Acc,
                              long long E, int F) {
    long long gid = (long long)blockIdx.x * blockDim.x + threadIdx.x;
    long long e   = gid >> 5;
    int lane      = (int)(gid & 31);
    if (e >= E) return;
    long long r = rows[e];
    long long cidx = cols[e];
    float nrm = dis[r] * dis[cidx];
    int per = F >> 5;                         // 4 (F=128) or 2 (F=64)
    const float* __restrict__ hs = Hsrc + (size_t)r * F + lane * per;
    float* __restrict__ ac = Acc + (size_t)cidx * F + lane * per;
    if (per == 4) {
        f4 hv = *(const f4*)hs;
#pragma unroll
        for (int j = 0; j < 4; ++j) atomicAdd(&ac[j], nrm * hv[j]);
    } else {
        f2 hv = *(const f2*)hs;
#pragma unroll
        for (int j = 0; j < 2; ++j) atomicAdd(&ac[j], nrm * hv[j]);
    }
}

// ---------------------------------------------------------------------------
// Finalize: Out += dis[i]^2 * Hsrc (self-loop) + bias; optional ReLU.
// ---------------------------------------------------------------------------
__global__ void finalize_layer(const float* __restrict__ Hsrc,
                               const float* __restrict__ dis,
                               const float* __restrict__ bias,
                               float* __restrict__ Out,
                               long long total, int F, int relu) {
    long long i = (long long)blockIdx.x * blockDim.x + threadIdx.x;
    if (i >= total) return;
    int node = (int)(i / F);
    int f    = (int)(i % F);
    float d = dis[node];
    float v = Out[i] + d * d * Hsrc[i] + bias[f];
    Out[i] = relu ? fmaxf(v, 0.0f) : v;
}

// ---------------------------------------------------------------------------
// Host-side orchestration
// ---------------------------------------------------------------------------
static inline int cdiv_ll(long long a, long long b) { return (int)((a + b - 1) / b); }

extern "C" void kernel_launch(void* const* d_in, const int* in_sizes, int n_in,
                              void* d_out, int out_size, void* d_ws, size_t ws_size,
                              hipStream_t stream) {
    const float*      x     = (const float*)d_in[0];
    const long long*  edges = (const long long*)d_in[1];
    const float*      W1    = (const float*)d_in[2];
    const float*      b1    = (const float*)d_in[3];
    const float*      W2    = (const float*)d_in[4];
    const float*      b2    = (const float*)d_in[5];
    const float*      W3    = (const float*)d_in[6];
    const float*      b3    = (const float*)d_in[7];
    float*            out   = (float*)d_out;

    const int F_IN  = 128;
    const int F_HID = 128;
    const int F_OUT = in_sizes[6] / F_HID;          // 64
    const int N     = in_sizes[0] / F_IN;            // 100000
    const long long E = (long long)in_sizes[1] / 2;  // 1600000

    const long long* rows = edges;
    const long long* cols = edges + E;

    // Workspace: dis[N] | bufA[N*128] | bufB[N*128] (f32)
    //            Xh,Xl [N*128] f16 | Wth,Wtl [128*128] f16
    float* ws   = (float*)d_ws;
    size_t Npad = ((size_t)N + 127) & ~(size_t)127;
    float* dis  = ws;
    float* bufA = ws + Npad;
    float* bufB = bufA + (size_t)N * F_HID;
    _Float16* Xh  = (_Float16*)(bufB + (size_t)N * F_HID);
    _Float16* Xl  = Xh + (size_t)N * F_HID;
    _Float16* Wth = Xl + (size_t)N * F_HID;
    _Float16* Wtl = Wth + (size_t)F_HID * F_HID;
    (void)ws_size; (void)n_in; (void)out_size;

    const int TB = 256;
    const long long NF  = (long long)N * F_HID;      // 12.8M
    const long long NFo = (long long)N * F_OUT;      // 6.4M
    const int splitGrid = cdiv_ll(NF / 8, TB);

    // --- normalization: dis = rsqrt(deg_col + 1) ---
    zero_f32<<<cdiv_ll(N, TB), TB, 0, stream>>>(dis, N);
    deg_count<<<cdiv_ll(E, TB), TB, 0, stream>>>(cols, dis, E);
    deg_finalize<<<cdiv_ll(N, TB), TB, 0, stream>>>(dis, N);

    dim3 gemmBlk(32);
    dim3 gemmGrid1(N / 16, F_HID / 16);
    dim3 gemmGrid3(N / 16, F_OUT / 16);
    int scatterGrid = cdiv_ll(E * 32, TB);

    // --- layer 1: h1 = relu(agg(x @ W1) + b1) ---
    split_f16x8<<<splitGrid, TB, 0, stream>>>(x, Xh, Xl, NF);
    wt_split_f16<<<cdiv_ll(F_HID * F_HID, TB), TB, 0, stream>>>(W1, Wth, Wtl, F_HID, F_HID);
    gcn_gemm_wmma<<<gemmGrid1, gemmBlk, 0, stream>>>(Xh, Xl, Wth, Wtl, bufA, F_HID);
    zero_f32<<<cdiv_ll(NF, TB), TB, 0, stream>>>(bufB, NF);
    scatter_edges<<<scatterGrid, TB, 0, stream>>>(rows, cols, dis, bufA, bufB, E, F_HID);
    finalize_layer<<<cdiv_ll(NF, TB), TB, 0, stream>>>(bufA, dis, b1, bufB, NF, F_HID, 1);

    // --- layer 2: h2 = relu(agg(h1 @ W2) + b2) ---
    split_f16x8<<<splitGrid, TB, 0, stream>>>(bufB, Xh, Xl, NF);
    wt_split_f16<<<cdiv_ll(F_HID * F_HID, TB), TB, 0, stream>>>(W2, Wth, Wtl, F_HID, F_HID);
    gcn_gemm_wmma<<<gemmGrid1, gemmBlk, 0, stream>>>(Xh, Xl, Wth, Wtl, bufA, F_HID);
    zero_f32<<<cdiv_ll(NF, TB), TB, 0, stream>>>(bufB, NF);
    scatter_edges<<<scatterGrid, TB, 0, stream>>>(rows, cols, dis, bufA, bufB, E, F_HID);
    finalize_layer<<<cdiv_ll(NF, TB), TB, 0, stream>>>(bufA, dis, b2, bufB, NF, F_HID, 1);

    // --- layer 3: out = agg(h2 @ W3) + b3 (no relu) ---
    split_f16x8<<<splitGrid, TB, 0, stream>>>(bufB, Xh, Xl, NF);
    wt_split_f16<<<cdiv_ll(F_HID * F_OUT, TB), TB, 0, stream>>>(W3, Wth, Wtl, F_HID, F_OUT);
    gcn_gemm_wmma<<<gemmGrid3, gemmBlk, 0, stream>>>(Xh, Xl, Wth, Wtl, bufA, F_OUT);
    zero_f32<<<cdiv_ll(NFo, TB), TB, 0, stream>>>(out, NFo);
    scatter_edges<<<scatterGrid, TB, 0, stream>>>(rows, cols, dis, bufA, out, E, F_OUT);
    finalize_layer<<<cdiv_ll(NFo, TB), TB, 0, stream>>>(bufA, dis, b3, out, NFo, F_OUT, 0);
}
